// Attention_28174985462461
// MI455X (gfx1250) — compile-verified, matrix operand order checked by reading, not run on hardware
//
#include <hip/hip_runtime.h>
#include <hip/hip_bf16.h>

#define B_ 8
#define S_ 2048
#define D_ 128

typedef __attribute__((ext_vector_type(16))) _Float16 v16h;
typedef __attribute__((ext_vector_type(4)))  _Float16 v4h;
typedef __attribute__((ext_vector_type(8)))  float    v8f;

#define NEG_BIG (-3.402823466e38f)

// Load one 16-bit A/B WMMA fragment (16x32 tile, this lane's slice) from a
// row-major f32 matrix. `p` must already point at row_start + kc + half*8.
// Layout (ISA 7.12.2, 16-bit A 16x32): lanes 0-15 hold M=0..15 with
// elements 0..7 = K {0..7}, elements 8..15 = K {16..23}; lanes 16-31 use
// K offsets +8 (folded into `p` by the caller).
__device__ inline v16h load_frag_f32(const float* __restrict__ p) {
  const float4* q = (const float4*)p;
  float4 x0 = q[0];       // k + 0..3
  float4 x1 = q[1];       // k + 4..7
  float4 y0 = q[4];       // k + 16..19
  float4 y1 = q[5];       // k + 20..23
  v16h r;
  r[0]  = (_Float16)x0.x; r[1]  = (_Float16)x0.y;
  r[2]  = (_Float16)x0.z; r[3]  = (_Float16)x0.w;
  r[4]  = (_Float16)x1.x; r[5]  = (_Float16)x1.y;
  r[6]  = (_Float16)x1.z; r[7]  = (_Float16)x1.w;
  r[8]  = (_Float16)y0.x; r[9]  = (_Float16)y0.y;
  r[10] = (_Float16)y0.z; r[11] = (_Float16)y0.w;
  r[12] = (_Float16)y1.x; r[13] = (_Float16)y1.y;
  r[14] = (_Float16)y1.z; r[15] = (_Float16)y1.w;
  return r;
}

// ---------------------------------------------------------------------------
// Kernel 1: attn = masked_softmax(Q K^T / sqrt(D) - |i-j|, mask)
// Grid: (S/16, B). Block: 256 (8 wave32). Wave w owns cols [w*256, w*256+256).
// Causal skip: tile cols jt..jt+15 can be non-zero only if jt <= i0+14.
// ---------------------------------------------------------------------------
__global__ __launch_bounds__(256) void attn_scores_kernel(
    const float* __restrict__ qm, const float* __restrict__ km,
    const float* __restrict__ rep, float* __restrict__ attn) {
  const int b   = blockIdx.y;
  const int i0  = blockIdx.x * 16;
  const int tid = threadIdx.x;
  const int w   = tid >> 5;
  const int ln  = tid & 31;
  const int lr  = ln & 15;   // row (A) / col (B) index within tile
  const int hf  = ln >> 4;   // which K-half this lane holds
  const int colbase = w * 256;
  const int jlim = i0 + 15;  // tile with first col jt participates iff jt < jlim

  __shared__ float redmax[8 * 16];
  __shared__ float redsum[8 * 16];

  v8f acc[16];
#pragma unroll
  for (int t = 0; t < 16; ++t) acc[t] = {};

  const bool wave_active = (colbase < jlim);

  const float* qrow = qm + ((size_t)b * S_ + i0 + lr) * D_ + hf * 8;
  const float* krow = km + ((size_t)b * S_ + colbase + lr) * D_ + hf * 8;

  if (wave_active) {
#pragma unroll
    for (int kc = 0; kc < D_; kc += 32) {
      v16h a = load_frag_f32(qrow + kc);
#pragma unroll
      for (int t = 0; t < 16; ++t) {
        if (colbase + t * 16 < jlim) {
          v16h bf = load_frag_f32(krow + (size_t)(t * 16) * D_ + kc);
          acc[t] = __builtin_amdgcn_wmma_f32_16x16x32_f16(
              false, a, false, bf, (short)0, acc[t], false, false);
        }
      }
    }
  }

  const float inv_scale = 0.08838834764831845f;  // 1/sqrt(128)

  float ri[8];
#pragma unroll
  for (int r = 0; r < 8; ++r) ri[r] = rep[(size_t)b * S_ + i0 + r + 8 * hf];

  // score + bias; masked slots become NEG_BIG so exp() underflows to 0 later.
  float lmax[8];
#pragma unroll
  for (int r = 0; r < 8; ++r) lmax[r] = NEG_BIG;
  if (wave_active) {
#pragma unroll
    for (int t = 0; t < 16; ++t) {
      if (colbase + t * 16 < jlim) {
        const int j = colbase + t * 16 + lr;
        const float rj = rep[(size_t)b * S_ + j];
#pragma unroll
        for (int r = 0; r < 8; ++r) {
          const int i = i0 + r + 8 * hf;
          const float s = acc[t][r] * inv_scale - fabsf((float)(i - j));
          const bool valid = (j < i) && (ri[r] != 0.0f) && (rj != 0.0f);
          const float sm = valid ? s : NEG_BIG;
          acc[t][r] = sm;
          lmax[r] = fmaxf(lmax[r], sm);
        }
      }
    }
  }
  // reduce within the 16-lane half (xor 1,2,4,8 stays inside the half)
#pragma unroll
  for (int r = 0; r < 8; ++r) {
#pragma unroll
    for (int m = 1; m <= 8; m <<= 1)
      lmax[r] = fmaxf(lmax[r], __shfl_xor(lmax[r], m, 32));
  }
  if (lr == 0) {
#pragma unroll
    for (int r = 0; r < 8; ++r) redmax[w * 16 + 8 * hf + r] = lmax[r];
  }
  __syncthreads();
  float rowmax[8];
#pragma unroll
  for (int r = 0; r < 8; ++r) {
    float mv = 0.0f;  // every row has >=1 masked slot (j>=i), contributing 0
#pragma unroll
    for (int ww = 0; ww < 8; ++ww) mv = fmaxf(mv, redmax[ww * 16 + 8 * hf + r]);
    rowmax[r] = mv;
  }

  // exp + local sum (masked slots hold NEG_BIG -> exp underflows to 0)
  float lsum[8];
#pragma unroll
  for (int r = 0; r < 8; ++r) lsum[r] = 0.0f;
  if (wave_active) {
#pragma unroll
    for (int t = 0; t < 16; ++t) {
      if (colbase + t * 16 < jlim) {
#pragma unroll
        for (int r = 0; r < 8; ++r) {
          const float e = __expf(acc[t][r] - rowmax[r]);
          acc[t][r] = e;
          lsum[r] += e;
        }
      } else {
#pragma unroll
        for (int r = 0; r < 8; ++r) acc[t][r] = 0.0f;  // clear raw acc in skipped tiles
      }
    }
  }
#pragma unroll
  for (int r = 0; r < 8; ++r) {
#pragma unroll
    for (int m = 1; m <= 8; m <<= 1)
      lsum[r] += __shfl_xor(lsum[r], m, 32);
  }
  if (lr == 0) {
#pragma unroll
    for (int r = 0; r < 8; ++r) redsum[w * 16 + 8 * hf + r] = lsum[r];
  }
  __syncthreads();
  float inv[8];
#pragma unroll
  for (int r = 0; r < 8; ++r) {
    float sm = 0.0f;
#pragma unroll
    for (int ww = 0; ww < 8; ++ww) sm += redsum[ww * 16 + 8 * hf + r];
    sm += (sm == 0.0f) ? 1.0f : 0.0f;      // reference: sums + (sums==0)
    inv[r] = 1.0f / (sm + 1e-20f);
  }

  float* ab = attn + (size_t)b * S_ * S_;
#pragma unroll
  for (int t = 0; t < 16; ++t) {
    const int j = colbase + t * 16 + lr;
#pragma unroll
    for (int r = 0; r < 8; ++r) {
      const int i = i0 + r + 8 * hf;
      ab[(size_t)i * S_ + j] = acc[t][r] * inv[r];
    }
  }
}

// ---------------------------------------------------------------------------
// Kernel 2: out = attn @ V.  Grid: (S/16, B). Block: 256 (8 wave32).
// Wave w owns output cols [w*16, w*16+16). V chunks double-buffered in LDS as
// f16. Causal bound: attn rows i0..i0+15 are zero for j >= i0+15, so the
// K-loop stops at i0+16 (halves average attn re-read traffic).
// ---------------------------------------------------------------------------
__device__ inline void load_vchunk(_Float16* __restrict__ dst,
                                   const float* __restrict__ vb,
                                   int kc, int tid) {
#pragma unroll
  for (int q4 = 0; q4 < 4; ++q4) {
    const int flat = (tid + q4 * 256) << 2;   // 0..4095, step 4
    const int row  = flat >> 7;
    const int col  = flat & 127;
    float4 x = *(const float4*)(vb + (size_t)(kc + row) * D_ + col);
    v4h h;
    h[0] = (_Float16)x.x; h[1] = (_Float16)x.y;
    h[2] = (_Float16)x.z; h[3] = (_Float16)x.w;
    *(v4h*)(dst + row * 128 + col) = h;
  }
}

__global__ __launch_bounds__(256) void attn_av_kernel(
    const float* __restrict__ attn, const float* __restrict__ vm,
    float* __restrict__ out) {
  const int b   = blockIdx.y;
  const int i0  = blockIdx.x * 16;
  const int tid = threadIdx.x;
  const int w   = tid >> 5;
  const int ln  = tid & 31;
  const int lr  = ln & 15;
  const int hf  = ln >> 4;

  __shared__ __align__(16) _Float16 vt[2][32 * 128];

  v8f acc = {};
  const float* arow = attn + ((size_t)b * S_ + i0 + lr) * S_ + hf * 8;
  const float* vb   = vm + (size_t)b * S_ * D_;
  const int n  = w * 16 + lr;
  const int kb = hf * 8;

  const int nchunks = (i0 + 16 + 31) / 32;   // causal bound: j < i0+16

  load_vchunk(&vt[0][0], vb, 0, tid);

  for (int c = 0; c < nchunks; ++c) {
    __syncthreads();   // vt[c&1] fully written by all waves
    if (c + 1 < nchunks) {
      load_vchunk(&vt[(c + 1) & 1][0], vb, (c + 1) * 32, tid);
      // hint the attn rows two chunks ahead (c+2) into cache
      __builtin_prefetch(arow + (c + 2) * 32, 0, 1);
    }
    const int kc = c * 32;
    v16h a = load_frag_f32(arow + kc);
    const _Float16* vbuf = &vt[c & 1][0];
    v16h bf;
#pragma unroll
    for (int jj = 0; jj < 8; ++jj) {
      bf[jj]     = vbuf[(kb + jj) * 128 + n];        // K = kb..kb+7
      bf[8 + jj] = vbuf[(16 + kb + jj) * 128 + n];   // K = 16+kb..16+kb+7
    }
    acc = __builtin_amdgcn_wmma_f32_16x16x32_f16(
        false, a, false, bf, (short)0, acc, false, false);
  }

#pragma unroll
  for (int r = 0; r < 8; ++r)
    out[((size_t)b * S_ + i0 + r + 8 * hf) * D_ + w * 16 + lr] = acc[r];
}

extern "C" void kernel_launch(void* const* d_in, const int* in_sizes, int n_in,
                              void* d_out, int out_size, void* d_ws, size_t ws_size,
                              hipStream_t stream) {
  (void)in_sizes; (void)n_in; (void)out_size; (void)d_ws; (void)ws_size;
  const float* q   = (const float*)d_in[0];
  const float* k   = (const float*)d_in[1];
  const float* v   = (const float*)d_in[2];
  const float* rep = (const float*)d_in[3];
  float* out  = (float*)d_out;
  float* attn = out + (size_t)B_ * S_ * D_;   // tuple: (out, attn) flat

  dim3 grid(S_ / 16, B_);
  dim3 blk(256);
  hipLaunchKernelGGL(attn_scores_kernel, grid, blk, 0, stream, q, k, rep, attn);
  hipLaunchKernelGGL(attn_av_kernel,     grid, blk, 0, stream, attn, v, out);
}
